// UnrolledSNU_27384711480033
// MI455X (gfx1250) — matrix operand order repaired and checked by reading, not compiled, not run on hardware
//
#include <hip/hip_runtime.h>
#include <hip/hip_bf16.h>
#include <math.h>

// Problem constants from the reference
#define BS 64
#define TT 256
#define FF 2048
#define UU 2048

typedef __attribute__((ext_vector_type(16))) __bf16 v16bf;
typedef __attribute__((ext_vector_type(8)))  __bf16 v8bf;
typedef __attribute__((ext_vector_type(2)))  __bf16 v2bf;
typedef __attribute__((ext_vector_type(8)))  float  v8f;

#define TM 128
#define TN 128
#define TK 32

// ---------------------------------------------------------------------------
// Async copy of 16 bytes global -> LDS, tracked by ASYNCcnt (CDNA5 path).
// Inline asm per CDNA5 ISA (portable across toolchains;
// the clang builtin's signature differs between clang-22 and clang-23).
// ---------------------------------------------------------------------------
__device__ __forceinline__ void async_b128(unsigned lds_addr, unsigned long long gaddr) {
    asm volatile("global_load_async_to_lds_b128 %0, %1, off"
                 :: "v"(lds_addr), "v"(gaddr) : "memory");
}
__device__ __forceinline__ void wait_async0() {
    asm volatile("s_wait_asynccnt 0x0" ::: "memory");
}

// ---------------------------------------------------------------------------
// Pre-pass A: Xb[M*K] = bf16(X[M*K])  (row-major, 8 elems/thread)
// ---------------------------------------------------------------------------
__global__ __launch_bounds__(256) void conv_x(const float* __restrict__ X,
                                              unsigned short* __restrict__ Xb,
                                              size_t n)
{
    size_t i = ((size_t)blockIdx.x * 256 + threadIdx.x) * 8;
    if (i >= n) return;
    float4 f0 = *(const float4*)(X + i);
    float4 f1 = *(const float4*)(X + i + 4);
    v8bf v;
    v[0] = (__bf16)f0.x; v[1] = (__bf16)f0.y; v[2] = (__bf16)f0.z; v[3] = (__bf16)f0.w;
    v[4] = (__bf16)f1.x; v[5] = (__bf16)f1.y; v[6] = (__bf16)f1.z; v[7] = (__bf16)f1.w;
    *(v8bf*)(Xb + i) = v;
}

// ---------------------------------------------------------------------------
// Pre-pass B: Wt[n][k] = bf16(W[k][n])  (32x32 LDS tile transpose)
// ---------------------------------------------------------------------------
__global__ __launch_bounds__(256) void conv_wt(const float* __restrict__ W,
                                               unsigned short* __restrict__ Wt,
                                               int K, int N)
{
    __shared__ __bf16 t[32][33];
    const int n0 = blockIdx.x * 32, k0 = blockIdx.y * 32;
    const int tx = threadIdx.x & 31, ty = threadIdx.x >> 5;   // ty: 0..7
    __bf16* Wtb = (__bf16*)Wt;
#pragma unroll
    for (int i = ty; i < 32; i += 8)
        t[i][tx] = (__bf16)W[(size_t)(k0 + i) * N + n0 + tx];
    __syncthreads();
#pragma unroll
    for (int i = ty; i < 32; i += 8)
        Wtb[(size_t)(n0 + i) * K + k0 + tx] = t[tx][i];
}

// ---------------------------------------------------------------------------
// Fast GEMM: Z[M,N] = Xb[M,K] * Wt[N,K]^T, bf16 in / f32 accumulate.
// 8 waves per block, block tile 128x128, wave tile 16x128 (8 WMMAs/k-step).
// Tiles move via global_load_async_to_lds_b128 into double-buffered LDS;
// copy of tile k+1 overlaps WMMA of tile k (one barrier per iteration).
// ---------------------------------------------------------------------------
__global__ __launch_bounds__(256) void snu_gemm_async(
    const unsigned short* __restrict__ Xb, const unsigned short* __restrict__ Wt,
    float* __restrict__ Z, int M, int N, int K)
{
    // buffer p: A tile [TM][TK] then B tile [TN][TK], both bf16 row-major
    __shared__ __align__(32) __bf16 smem[2][(TM + TN) * TK];

    const int tid   = threadIdx.x;
    const int lane  = tid & 31;
    const int wave  = tid >> 5;
    const int laneM = lane & 15;
    const int half  = lane >> 4;

    const int bm = blockIdx.x * TM;
    const int bn = blockIdx.y * TN;

    v8f acc[8];
#pragma unroll
    for (int i = 0; i < 8; ++i) acc[i] = (v8f){0.f,0.f,0.f,0.f,0.f,0.f,0.f,0.f};

    // copy slots: tile row = 64B, 4 lanes/row, 2 rows/thread per matrix
    const int s_row0 = tid >> 2;           // 0..63
    const int s_off  = (tid & 3) * 16;     // 0,16,32,48 bytes within row

    const unsigned long long xrow =
        (unsigned long long)(const void*)Xb + ((size_t)bm * K) * 2;
    const unsigned long long wrow =
        (unsigned long long)(const void*)Wt + ((size_t)bn * K) * 2;
    const size_t gstride = (size_t)K * 2;  // bytes per matrix row

    auto issue_tile = [&](int k0, int p) {
        const unsigned aBase = (unsigned)(size_t)&smem[p][0];
        const unsigned bBase = aBase + TM * TK * 2;
        const size_t kb = (size_t)k0 * 2 + s_off;
#pragma unroll
        for (int r = 0; r < 2; ++r) {
            const int row = s_row0 + r * 64;
            async_b128(aBase + row * (TK * 2) + s_off, xrow + (size_t)row * gstride + kb);
            async_b128(bBase + row * (TK * 2) + s_off, wrow + (size_t)row * gstride + kb);
        }
    };

    issue_tile(0, 0);
    int p = 0;
    for (int k0 = 0; k0 < K; k0 += TK) {
        wait_async0();          // my async writes into smem[p] have landed
        __syncthreads();        // everyone's have (each waited before arriving)
        if (k0 + TK < K) issue_tile(k0 + TK, p ^ 1);   // overlaps compute below

        const __bf16* lA = &smem[p][0];
        const __bf16* lB = &smem[p][TM * TK];

        // A fragment (16x32): lane = row laneM of wave strip,
        // halfwords = K half*16..+15
        const int arow = wave * 16 + laneM;
        v8bf a0 = *(const v8bf*)(lA + arow * TK + half * 16);
        v8bf a1 = *(const v8bf*)(lA + arow * TK + half * 16 + 8);
        v16bf Af = __builtin_shufflevector(a0, a1,
                     0,1,2,3,4,5,6,7,8,9,10,11,12,13,14,15);

        // 8 B fragments: lane = col laneM, halfwords = K half*16..+15
#pragma unroll
        for (int nt = 0; nt < 8; ++nt) {
            v16bf Bf = *(const v16bf*)(lB + (nt * 16 + laneM) * TK + half * 16);
            acc[nt] = __builtin_amdgcn_wmma_f32_16x16x32_bf16(
                false, Af, false, Bf, (short)0, acc[nt], false, false);
        }
        p ^= 1;
    }

    // C store: lanes 0-15 -> M=r, lanes 16-31 -> M=8+r; N = laneM
#pragma unroll
    for (int nt = 0; nt < 8; ++nt) {
        const int col = bn + nt * 16 + laneM;
#pragma unroll
        for (int r = 0; r < 8; ++r) {
            const int row = bm + wave * 16 + half * 8 + r;
            Z[(size_t)row * N + col] = acc[nt][r];
        }
    }
}

// ---------------------------------------------------------------------------
// Fallback GEMM (scratch too small for bf16 staging): convert in-loop.
// ---------------------------------------------------------------------------
__global__ __launch_bounds__(256) void snu_gemm_bf16(
    const float* __restrict__ X, const float* __restrict__ W,
    float* __restrict__ Z, int M, int N, int K)
{
    __shared__ __align__(32) __bf16 lA[TM][TK];
    __shared__ __align__(32) __bf16 lB[TN][TK];

    const int tid   = threadIdx.x;
    const int lane  = tid & 31;
    const int wave  = tid >> 5;
    const int laneM = lane & 15;
    const int half  = lane >> 4;
    const int bm = blockIdx.x * TM;
    const int bn = blockIdx.y * TN;

    v8f acc[8];
#pragma unroll
    for (int i = 0; i < 8; ++i) acc[i] = (v8f){0.f,0.f,0.f,0.f,0.f,0.f,0.f,0.f};

    const int a_row = tid >> 2;
    const int a_seg = (tid & 3) * 8;
    const int b_kk  = (tid >> 4) * 2;
    const int b_ns  = (tid & 15) * 8;

    for (int k0 = 0; k0 < K; k0 += TK) {
        __syncthreads();
#pragma unroll
        for (int pp = 0; pp < 2; ++pp) {
            const int r = a_row + pp * 64;
            const float* src = X + (size_t)(bm + r) * K + (k0 + a_seg);
            float4 f0 = *(const float4*)(src);
            float4 f1 = *(const float4*)(src + 4);
            v8bf v;
            v[0] = (__bf16)f0.x; v[1] = (__bf16)f0.y;
            v[2] = (__bf16)f0.z; v[3] = (__bf16)f0.w;
            v[4] = (__bf16)f1.x; v[5] = (__bf16)f1.y;
            v[6] = (__bf16)f1.z; v[7] = (__bf16)f1.w;
            *(v8bf*)(&lA[r][a_seg]) = v;
        }
        {
            const float* s0 = W + (size_t)(k0 + b_kk) * N + (bn + b_ns);
            const float* s1 = s0 + N;
            float4 a0 = *(const float4*)(s0), a1 = *(const float4*)(s0 + 4);
            float4 c0 = *(const float4*)(s1), c1 = *(const float4*)(s1 + 4);
            float lo[8] = {a0.x,a0.y,a0.z,a0.w,a1.x,a1.y,a1.z,a1.w};
            float hi[8] = {c0.x,c0.y,c0.z,c0.w,c1.x,c1.y,c1.z,c1.w};
#pragma unroll
            for (int i = 0; i < 8; ++i) {
                v2bf pr; pr[0] = (__bf16)lo[i]; pr[1] = (__bf16)hi[i];
                *(v2bf*)(&lB[b_ns + i][b_kk]) = pr;
            }
        }
        __syncthreads();

        const int arow = wave * 16 + laneM;
        v8bf a0 = *(const v8bf*)(&lA[arow][half * 16]);
        v8bf a1 = *(const v8bf*)(&lA[arow][half * 16 + 8]);
        v16bf Af = __builtin_shufflevector(a0, a1,
                     0,1,2,3,4,5,6,7,8,9,10,11,12,13,14,15);
#pragma unroll
        for (int nt = 0; nt < 8; ++nt) {
            v16bf Bf = *(const v16bf*)(&lB[nt * 16 + laneM][half * 16]);
            acc[nt] = __builtin_amdgcn_wmma_f32_16x16x32_bf16(
                false, Af, false, Bf, (short)0, acc[nt], false, false);
        }
    }
#pragma unroll
    for (int nt = 0; nt < 8; ++nt) {
        const int col = bn + nt * 16 + laneM;
#pragma unroll
        for (int r = 0; r < 8; ++r) {
            const int row = bm + wave * 16 + half * 8 + r;
            Z[(size_t)row * N + col] = acc[nt][r];
        }
    }
}

// ---------------------------------------------------------------------------
// Sequential SNU scan over T: one thread per (b,u), coalesced stride-U.
// ---------------------------------------------------------------------------
__global__ __launch_bounds__(256) void snu_scan(
    const float* __restrict__ Z, const float* __restrict__ bias,
    const float* __restrict__ decay, float* __restrict__ Y)
{
    const int idx = blockIdx.x * blockDim.x + threadIdx.x;
    if (idx >= BS * UU) return;
    const int u = idx % UU;
    const int b = idx / UU;

    const float d  = decay[0];
    const float bu = bias[u];

    float Vm = 0.f, h = 0.f;
    size_t base = (size_t)b * TT * UU + u;
#pragma unroll 4
    for (int t = 0; t < TT; ++t) {
        const float z = Z[base];
        Vm = fmaxf(z + d * Vm * (1.f - h), 0.f);
        h  = 1.f / (1.f + __expf(-(Vm + bu)));
        Y[base] = h;
        base += UU;
    }
}

// ---------------------------------------------------------------------------
extern "C" void kernel_launch(void* const* d_in, const int* in_sizes, int n_in,
                              void* d_out, int out_size, void* d_ws, size_t ws_size,
                              hipStream_t stream)
{
    const float* x     = (const float*)d_in[0];  // [B,T,F]
    const float* W     = (const float*)d_in[1];  // [F,U]
    const float* bias  = (const float*)d_in[2];  // [U]
    const float* decay = (const float*)d_in[3];  // [1]
    float* out = (float*)d_out;                  // [B,T,U]

    const int M = BS * TT, N = UU, K = FF;
    const size_t zBytes = (size_t)M * N * sizeof(float);    // 134 MB
    const size_t xBytes = (size_t)M * K * 2;                // 67 MB
    const size_t wBytes = (size_t)K * N * 2;                // 8.4 MB

    float* Z = (float*)d_ws;
    dim3 ggrid(M / TM, N / TN);

    if (ws_size >= zBytes + xBytes + wBytes) {
        unsigned short* Xb = (unsigned short*)((char*)d_ws + zBytes);
        unsigned short* Wt = (unsigned short*)((char*)d_ws + zBytes + xBytes);
        const size_t nx = (size_t)M * K;
        conv_x<<<(unsigned)(nx / (256 * 8)), 256, 0, stream>>>(x, Xb, nx);
        conv_wt<<<dim3(N / 32, K / 32), 256, 0, stream>>>(W, Wt, K, N);
        snu_gemm_async<<<ggrid, 256, 0, stream>>>(Xb, Wt, Z, M, N, K);
    } else {
        snu_gemm_bf16<<<ggrid, 256, 0, stream>>>(x, W, Z, M, N, K);
    }

    const int total = BS * UU;
    snu_scan<<<(total + 255) / 256, 256, 0, stream>>>(Z, bias, decay, out);
}
